// MultiHeadAttention_23192823398678
// MI455X (gfx1250) — matrix operand order repaired and checked by reading, not compiled, not run on hardware
//
#include <hip/hip_runtime.h>
#include <hip/hip_bf16.h>

typedef __attribute__((ext_vector_type(16))) _Float16 v16h;
typedef __attribute__((ext_vector_type(8)))  _Float16 v8h_t;
typedef __attribute__((ext_vector_type(8)))  float    v8f;
typedef __attribute__((ext_vector_type(4)))  float    f32x4;

// ---- CDNA5 async-to-LDS helpers (ASYNCcnt-tracked DMA, ISA ch.10/15.18) ----
// Copies 16 bytes per lane from global memory directly into LDS, bypassing
// VGPRs.  LDS operand is the wave-relative LDS byte address (= low 32 bits of
// the generic flat pointer per the LDS aperture mapping).
__device__ __forceinline__ void asyncLdsCopyB128(void* ldsPtr, const void* gPtr) {
    unsigned int loff = (unsigned int)(unsigned long long)ldsPtr;
    asm volatile("global_load_async_to_lds_b128 %0, %1, off"
                 :: "v"(loff), "v"(gPtr) : "memory");
}
__device__ __forceinline__ void waitAsync0() {
    asm volatile("s_wait_asynccnt 0" ::: "memory");
}

// ---- WMMA fragment loaders -------------------------------------------------
// A operand (16x32 f16, M x K): lane holds row (lane&15); halves 0..7 = K kA..kA+7,
// halves 8..15 = K kA+16..kA+23, where kA = (lane>>4)*8.
__device__ __forceinline__ v16h loadFragA(const _Float16* p) {
    v8h_t lo = *(const v8h_t*)(p);
    v8h_t hi = *(const v8h_t*)(p + 16);
    v16h r;
#pragma unroll
    for (int i = 0; i < 8; ++i) { r[i] = lo[i]; r[i + 8] = hi[i]; }
    return r;
}

// B operand (32x16 f16, K x N): lane holds column (lane&15); halves 0..7 = K kB..kB+7,
// halves 8..15 = K kB+16..kB+23, where kB = (lane>>4)*8.
__device__ __forceinline__ v16h loadFragB(const _Float16* p) {
    v8h_t lo = *(const v8h_t*)(p);
    v8h_t hi = *(const v8h_t*)(p + 16);
    v16h r;
#pragma unroll
    for (int i = 0; i < 8; ++i) { r[i] = lo[i]; r[i + 8] = hi[i]; }
    return r;
}

#define WMMA_F32_F16(a, b, c) \
    __builtin_amdgcn_wmma_f32_16x16x32_f16(false, (a), false, (b), (short)0, (c), false, false)

// ---- Kernel 1: projection GEMM  C = x(f32->f16) @ W(f32->f16) + b ----------
// Out written as f16 in head-split layout [B, H, S, 64].  scale folded in (for Q).
__global__ __launch_bounds__(256) void mha_gemm_proj(
    const float* __restrict__ A,    // [8192, 1024] row-major (x)
    const float* __restrict__ W,    // [1024, 1024] (in, out)
    const float* __restrict__ bias, // [1024]
    _Float16* __restrict__ Out,     // [B,H,S,64]
    float scale)
{
    constexpr int K = 1024, N = 1024, SEQ = 2048, LDA = 40;
    __shared__ _Float16 As[128 * LDA];   // x tile   [m][k]
    __shared__ _Float16 Bs[128 * LDA];   // W tile transposed: [n][k]

    const int tid  = threadIdx.x;
    const int lane = tid & 31;
    const int wave = tid >> 5;
    const int waveM = wave >> 1;   // 0..3  (64 rows each)
    const int waveN = wave & 1;    // 0..1  (32 cols each)
    const int m0 = blockIdx.y * 128;
    const int n0 = blockIdx.x * 128;

    v8f acc[4][2] = {};

    const int kA  = (lane >> 4) << 3;   // A-frag k offset
    const int kB  = (lane >> 4) << 3;   // B-frag k offset

    for (int k0 = 0; k0 < K; k0 += 32) {
        // stage A tile (fp32 -> f16): 128x32, 16 elems/thread
        {
            int row = tid >> 1, c0 = (tid & 1) * 16;
            const float* g = A + (size_t)(m0 + row) * K + k0 + c0;
            f32x4 x0 = *(const f32x4*)(g);
            f32x4 x1 = *(const f32x4*)(g + 4);
            f32x4 x2 = *(const f32x4*)(g + 8);
            f32x4 x3 = *(const f32x4*)(g + 12);
            _Float16* s = As + row * LDA + c0;
#pragma unroll
            for (int i = 0; i < 4; ++i) {
                s[i]      = (_Float16)x0[i];
                s[4 + i]  = (_Float16)x1[i];
                s[8 + i]  = (_Float16)x2[i];
                s[12 + i] = (_Float16)x3[i];
            }
            if (k0 + 32 < K) __builtin_prefetch((const void*)(g + 32), 0, 1);
        }
        // stage W tile transposed (fp32 -> f16): Bs[n][k]
        {
            int kk = tid & 31, nc = (tid >> 5) * 16;
            const float* g = W + (size_t)(k0 + kk) * N + n0 + nc;
#pragma unroll
            for (int i = 0; i < 16; ++i)
                Bs[(nc + i) * LDA + kk] = (_Float16)g[i];
            if (k0 + 32 < K) __builtin_prefetch((const void*)(g + 32 * N), 0, 1);
        }
        __syncthreads();

        v16h afrag[4];
#pragma unroll
        for (int wm = 0; wm < 4; ++wm) {
            int row = waveM * 64 + wm * 16 + (lane & 15);
            afrag[wm] = loadFragA(As + row * LDA + kA);
        }
#pragma unroll
        for (int wn = 0; wn < 2; ++wn) {
            int nrow = waveN * 32 + wn * 16 + (lane & 15);
            v16h bfrag = loadFragB(Bs + nrow * LDA + kB);
#pragma unroll
            for (int wm = 0; wm < 4; ++wm)
                acc[wm][wn] = WMMA_F32_F16(afrag[wm], bfrag, acc[wm][wn]);
        }
        __syncthreads();
    }

    // epilogue: +bias, *scale, write f16 head-split [B,H,S,64]
#pragma unroll
    for (int wm = 0; wm < 4; ++wm) {
        int mbase = m0 + waveM * 64 + wm * 16 + ((lane >> 4) << 3);
#pragma unroll
        for (int wn = 0; wn < 2; ++wn) {
            int n = n0 + waveN * 32 + wn * 16 + (lane & 15);
            int h = n >> 6, d = n & 63;
            float bv = bias[n];
#pragma unroll
            for (int v = 0; v < 8; ++v) {
                int m = mbase + v;
                int b = m >> 11, s = m & (SEQ - 1);
                float val = (acc[wm][wn][v] + bv) * scale;
                Out[(((size_t)(b * 16 + h) * SEQ + s) << 6) + d] = (_Float16)val;
            }
        }
    }
}

// ---- Kernel 2: flash attention over causal mask ----------------------------
// One workgroup (8 waves) handles 128 query rows of one (b,h).
__global__ __launch_bounds__(256) void mha_attn(
    const _Float16* __restrict__ Q,  // [B,H,S,64], pre-scaled by 1/sqrt(dk)
    const _Float16* __restrict__ Kg, // [B,H,S,64]
    const _Float16* __restrict__ Vg, // [B,H,S,64]
    _Float16* __restrict__ Og)       // [B,S,H,64]  (= [8192,1024] for out proj)
{
    constexpr int SEQ = 2048, DK = 64, LDK = 72;
    __shared__ _Float16 Ksh[64 * LDK];       // K rows  [key][d]
    __shared__ _Float16 Vt [64 * LDK];       // V^T     [d][key]
    __shared__ _Float16 Psh[8 * 16 * LDK];   // per-wave P tile [16][64]

    const int tid  = threadIdx.x;
    const int lane = tid & 31;
    const int wave = tid >> 5;
    const int q0   = blockIdx.x * 128;
    const int bh   = blockIdx.y;
    const size_t headOff = (size_t)bh * SEQ * DK;
    const _Float16* Qh = Q  + headOff;
    const _Float16* Kh = Kg + headOff;
    const _Float16* Vh = Vg + headOff;

    const int kA = (lane >> 4) << 3;
    const int kB = (lane >> 4) << 3;
    const int qrow = q0 + wave * 16;

    // Q fragments held in registers for the whole kernel
    v16h qf[2];
    {
        const _Float16* p = Qh + (size_t)(qrow + (lane & 15)) * DK + kA;
        qf[0] = loadFragA(p);
        qf[1] = loadFragA(p + 32);
    }

    v8f o[4] = {};
    float m_i[8], l_i[8];
#pragma unroll
    for (int v = 0; v < 8; ++v) { m_i[v] = -3.0e38f; l_i[v] = 0.0f; }

    _Float16* Pw = Psh + wave * 16 * LDK;
    const int nkb = (q0 + 128) / 64;   // causal: key blocks covering keys <= q0+127

    for (int kb = 0; kb < nkb; ++kb) {
        const int k0 = kb * 64;
        __syncthreads();   // previous iteration finished reading Ksh/Vt
        // stage K rows via CDNA5 async-to-LDS DMA (straight f16 copy),
        // stage V transposed through VGPRs (layout change), 16 elems/thread each
        {
            int key = tid >> 2, c0 = (tid & 3) * 16;
            const _Float16* gk = Kh + (size_t)(k0 + key) * DK + c0;
            _Float16* s = Ksh + key * LDK + c0;
            asyncLdsCopyB128((void*)(s),     (const void*)(gk));
            asyncLdsCopyB128((void*)(s + 8), (const void*)(gk + 8));
            const _Float16* gv = Vh + (size_t)(k0 + key) * DK + c0;
#pragma unroll
            for (int i = 0; i < 16; ++i)
                Vt[(c0 + i) * LDK + key] = gv[i];
        }
        waitAsync0();
        __syncthreads();

        // scores S = Q * K^T  (16 x 64 per wave)
        v8f sacc[4] = {};
#pragma unroll
        for (int kn = 0; kn < 4; ++kn) {
            const _Float16* kp = Ksh + (kn * 16 + (lane & 15)) * LDK;
            v16h b0 = loadFragB(kp + kB);
            v16h b1 = loadFragB(kp + 32 + kB);
            sacc[kn] = WMMA_F32_F16(qf[0], b0, sacc[kn]);
            sacc[kn] = WMMA_F32_F16(qf[1], b1, sacc[kn]);
        }

        // causal mask + online softmax
        const int rbase = qrow + ((lane >> 4) << 3);
        const int cbase = k0 + (lane & 15);
#pragma unroll
        for (int v = 0; v < 8; ++v) {
            int q = rbase + v;
            float mx = -3.0e38f;
#pragma unroll
            for (int kn = 0; kn < 4; ++kn) {
                float sv = sacc[kn][v];
                sv = ((cbase + kn * 16) > q) ? -1.0e9f : sv;
                sacc[kn][v] = sv;
                mx = fmaxf(mx, sv);
            }
#pragma unroll
            for (int xm = 1; xm < 16; xm <<= 1)
                mx = fmaxf(mx, __shfl_xor(mx, xm, 32));
            float mn = fmaxf(m_i[v], mx);
            float alpha = __expf(m_i[v] - mn);
            float rs = 0.0f;
#pragma unroll
            for (int kn = 0; kn < 4; ++kn) {
                float p = __expf(sacc[kn][v] - mn);
                sacc[kn][v] = p;
                rs += p;
            }
#pragma unroll
            for (int xm = 1; xm < 16; xm <<= 1)
                rs += __shfl_xor(rs, xm, 32);
            l_i[v] = l_i[v] * alpha + rs;
            m_i[v] = mn;
#pragma unroll
            for (int dn = 0; dn < 4; ++dn)
                o[dn][v] *= alpha;
        }

        // P (C-layout) -> LDS -> reload in A-layout
#pragma unroll
        for (int v = 0; v < 8; ++v) {
            int rl = v + ((lane >> 4) << 3);
#pragma unroll
            for (int kn = 0; kn < 4; ++kn)
                Pw[rl * LDK + kn * 16 + (lane & 15)] = (_Float16)sacc[kn][v];
        }
        __syncthreads();

        v16h pf[2];
        {
            const _Float16* p = Pw + (lane & 15) * LDK + kA;
            pf[0] = loadFragA(p);
            pf[1] = loadFragA(p + 32);
        }
#pragma unroll
        for (int dn = 0; dn < 4; ++dn) {
            const _Float16* vp = Vt + (dn * 16 + (lane & 15)) * LDK;
            v16h v0 = loadFragB(vp + kB);
            v16h v1 = loadFragB(vp + 32 + kB);
            o[dn] = WMMA_F32_F16(pf[0], v0, o[dn]);
            o[dn] = WMMA_F32_F16(pf[1], v1, o[dn]);
        }
    }

    // finalize: O / l, write f16 [B,S,H,64]
    const int b = bh >> 4, h = bh & 15;
#pragma unroll
    for (int v = 0; v < 8; ++v) {
        float inv = 1.0f / l_i[v];
        int s = qrow + v + ((lane >> 4) << 3);
#pragma unroll
        for (int dn = 0; dn < 4; ++dn) {
            int d = dn * 16 + (lane & 15);
            Og[(((size_t)(b * SEQ + s) * 16 + h) << 6) + d] = (_Float16)(o[dn][v] * inv);
        }
    }
}

// ---- Kernel 3: output projection  out = attn(f16) @ Wo(f32->f16) + bo ------
__global__ __launch_bounds__(256) void mha_gemm_out(
    const _Float16* __restrict__ A, // [8192, 1024] f16 (attn, heads concatenated)
    const float* __restrict__ W,    // [1024, 1024]
    const float* __restrict__ bias, // [1024]
    float* __restrict__ Out)        // [8192, 1024] fp32
{
    constexpr int K = 1024, N = 1024, LDA = 40;
    __shared__ _Float16 As[128 * LDA];
    __shared__ _Float16 Bs[128 * LDA];

    const int tid  = threadIdx.x;
    const int lane = tid & 31;
    const int wave = tid >> 5;
    const int waveM = wave >> 1;
    const int waveN = wave & 1;
    const int m0 = blockIdx.y * 128;
    const int n0 = blockIdx.x * 128;

    v8f acc[4][2] = {};
    const int kA = (lane >> 4) << 3;
    const int kB = (lane >> 4) << 3;

    for (int k0 = 0; k0 < K; k0 += 32) {
        // stage A tile via async-to-LDS DMA (straight f16 copy, 32B/thread)
        {
            int row = tid >> 1, c0 = (tid & 1) * 16;
            const _Float16* g = A + (size_t)(m0 + row) * K + k0 + c0;
            _Float16* s = As + row * LDA + c0;
            asyncLdsCopyB128((void*)(s),     (const void*)(g));
            asyncLdsCopyB128((void*)(s + 8), (const void*)(g + 8));
            if (k0 + 32 < K) __builtin_prefetch((const void*)(g + 32), 0, 1);
        }
        // stage W tile transposed (fp32 -> f16): Bs[n][k]
        {
            int kk = tid & 31, nc = (tid >> 5) * 16;
            const float* g = W + (size_t)(k0 + kk) * N + n0 + nc;
#pragma unroll
            for (int i = 0; i < 16; ++i)
                Bs[(nc + i) * LDA + kk] = (_Float16)g[i];
            if (k0 + 32 < K) __builtin_prefetch((const void*)(g + 32 * N), 0, 1);
        }
        waitAsync0();
        __syncthreads();

        v16h afrag[4];
#pragma unroll
        for (int wm = 0; wm < 4; ++wm) {
            int row = waveM * 64 + wm * 16 + (lane & 15);
            afrag[wm] = loadFragA(As + row * LDA + kA);
        }
#pragma unroll
        for (int wn = 0; wn < 2; ++wn) {
            int nrow = waveN * 32 + wn * 16 + (lane & 15);
            v16h bfrag = loadFragB(Bs + nrow * LDA + kB);
#pragma unroll
            for (int wm = 0; wm < 4; ++wm)
                acc[wm][wn] = WMMA_F32_F16(afrag[wm], bfrag, acc[wm][wn]);
        }
        __syncthreads();
    }

#pragma unroll
    for (int wm = 0; wm < 4; ++wm) {
        int mbase = m0 + waveM * 64 + wm * 16 + ((lane >> 4) << 3);
#pragma unroll
        for (int wn = 0; wn < 2; ++wn) {
            int n = n0 + waveN * 32 + wn * 16 + (lane & 15);
            float bv = bias[n];
#pragma unroll
            for (int v = 0; v < 8; ++v) {
                int m = mbase + v;
                Out[(size_t)m * N + n] = acc[wm][wn][v] + bv;
            }
        }
    }
}

// ---- launch ---------------------------------------------------------------
extern "C" void kernel_launch(void* const* d_in, const int* in_sizes, int n_in,
                              void* d_out, int out_size, void* d_ws, size_t ws_size,
                              hipStream_t stream) {
    const float* x  = (const float*)d_in[0];
    // d_in[1] = causal mask (bool) -- computed analytically in-kernel
    const float* Wq = (const float*)d_in[2];
    const float* bq = (const float*)d_in[3];
    const float* Wk = (const float*)d_in[4];
    const float* bk = (const float*)d_in[5];
    const float* Wv = (const float*)d_in[6];
    const float* bv = (const float*)d_in[7];
    const float* Wo = (const float*)d_in[8];
    const float* bo = (const float*)d_in[9];
    float* out = (float*)d_out;

    const size_t MAT = (size_t)8192 * 1024 * sizeof(_Float16);  // 16 MB each
    char* ws = (char*)d_ws;
    _Float16* Qh = (_Float16*)(ws);
    _Float16* Kh = (_Float16*)(ws + MAT);
    _Float16* Vh = (_Float16*)(ws + 2 * MAT);
    _Float16* Ah = (_Float16*)(ws + 3 * MAT);

    dim3 blk(256);
    dim3 gGemm(8, 64);   // N/128, M/128
    const float SCALE = 0.125f;  // 1/sqrt(64)

    mha_gemm_proj<<<gGemm, blk, 0, stream>>>(x, Wq, bq, Qh, SCALE);
    mha_gemm_proj<<<gGemm, blk, 0, stream>>>(x, Wk, bk, Kh, 1.0f);
    mha_gemm_proj<<<gGemm, blk, 0, stream>>>(x, Wv, bv, Vh, 1.0f);
    mha_attn<<<dim3(16, 64), blk, 0, stream>>>(Qh, Kh, Vh, Ah);
    mha_gemm_out<<<gGemm, blk, 0, stream>>>(Ah, Wo, bo, out);
}
